// MultiHeadAttention_25254407701262
// MI455X (gfx1250) — matrix operand order repaired and checked by reading, not compiled, not run on hardware
//
#include <hip/hip_runtime.h>
#include <hip/hip_bf16.h>
#include <stdint.h>

typedef __attribute__((ext_vector_type(16))) __bf16 v16bf;
typedef __attribute__((ext_vector_type(16))) unsigned short v16u;
typedef __attribute__((ext_vector_type(8)))  float v8f;

#define D_MODEL 1024
#define N_HEADS 16
#define DK      64
#define BATCH   4
#define SEQ     2048
#define NTOK    (BATCH * SEQ)

// ---------- helpers ----------

__device__ __forceinline__ unsigned short f2bf(float f) {
    unsigned int u = __float_as_uint(f);
    u += 0x7FFFu + ((u >> 16) & 1u);   // round-to-nearest-even
    return (unsigned short)(u >> 16);
}

struct U32x8 { uint4 a, b; };

__device__ __forceinline__ v16bf ld_frag(const unsigned short* p) {
    U32x8 t;
    t.a = *(const uint4*)(p);
    t.b = *(const uint4*)(p + 8);
    return __builtin_bit_cast(v16bf, t);
}

__device__ __forceinline__ v16bf ld_frag2(const unsigned short* p0,
                                          const unsigned short* p1) {
    U32x8 t;
    t.a = *(const uint4*)(p0);
    t.b = *(const uint4*)(p1);
    return __builtin_bit_cast(v16bf, t);
}

__device__ __forceinline__ v8f zero8() {
    v8f z;
#pragma unroll
    for (int i = 0; i < 8; i++) z[i] = 0.0f;
    return z;
}

// 16-byte async global->LDS copy (ASYNCcnt-tracked, no VGPR data path)
__device__ __forceinline__ void async_cp16(const unsigned short* lds_dst,
                                           const unsigned short* gsrc) {
    unsigned lds = (unsigned)(uintptr_t)lds_dst;
    unsigned long long ga = (unsigned long long)(uintptr_t)gsrc;
    asm volatile("global_load_async_to_lds_b128 %0, %1, off"
                 :: "v"(lds), "v"(ga) : "memory");
}
template <int N>
__device__ __forceinline__ void wait_async() {
    asm volatile("s_wait_asynccnt %0" :: "n"(N) : "memory");
}
__device__ __forceinline__ void wait_ds0() {
    asm volatile("s_wait_dscnt 0x0" ::: "memory");
}

// ---------- pre-pass: f32 -> bf16 convert ----------

__global__ void __launch_bounds__(256)
cvt_bf16(const float* __restrict__ in, unsigned short* __restrict__ out, int n) {
    int i = (blockIdx.x * 256 + threadIdx.x) * 4;
    if (i < n) {
        float4 v = *(const float4*)(in + i);
        ushort4 u;
        u.x = f2bf(v.x); u.y = f2bf(v.y); u.z = f2bf(v.z); u.w = f2bf(v.w);
        *(ushort4*)(out + i) = u;
    }
}

// ---------- pre-pass: Wt[k][n] = bf16(W[n][k]) ----------

__global__ void __launch_bounds__(256)
transpose_cvt(const float* __restrict__ W, unsigned short* __restrict__ Wt) {
    __shared__ float tile[32][33];
    const int n0 = blockIdx.y * 32, k0 = blockIdx.x * 32;
    const int col = threadIdx.x & 31, row = threadIdx.x >> 5;  // 32 x 8
#pragma unroll
    for (int i = 0; i < 4; i++)
        tile[row + 8 * i][col] = W[(size_t)(n0 + row + 8 * i) * D_MODEL + k0 + col];
    __syncthreads();
#pragma unroll
    for (int i = 0; i < 4; i++)
        Wt[(size_t)(k0 + row + 8 * i) * D_MODEL + n0 + col] =
            f2bf(tile[col][row + 8 * i]);
}

// ---------- GEMM: C[M,1024] = A[M,1024](bf16) @ Wt(bf16,[k][n]) + bias ----
// Block: 256 threads = 8 waves (4 M x 2 N). Block tile 128x128, BK=32.
// Double-buffered async global->LDS staging pipelined against WMMA.

template <bool OUT_BF16>
__global__ void __launch_bounds__(256)
gemm_bias(const unsigned short* __restrict__ A,
          const unsigned short* __restrict__ Wt,
          const float* __restrict__ bias, void* __restrict__ Optr) {
    __shared__ __align__(16) unsigned short As[2][128 * 40];  // pad 32->40
    __shared__ __align__(16) unsigned short Bs[2][32 * 136];  // pad 128->136

    const int tid  = threadIdx.x;
    const int lane = tid & 31;
    const int wave = tid >> 5;
    const int wm   = wave & 3;   // 0..3
    const int wn   = wave >> 2;  // 0..1
    const int m0   = blockIdx.y * 128;
    const int n0   = blockIdx.x * 128;

    v8f c[2][4];
#pragma unroll
    for (int i = 0; i < 2; i++)
#pragma unroll
        for (int j = 0; j < 4; j++) c[i][j] = zero8();

    const int mlo = lane & 15;
    const int hi  = lane >> 4;

    // per-thread staging coordinates
    const int arow = tid >> 1, ach = (tid & 1) * 16;       // A: 128 rows x 32
    const int bk   = tid >> 3, bc  = (tid & 7) * 16;       // B: 32 rows x 128

    auto stage = [&](int k0, int bi) {
        const unsigned short* gA = A + (size_t)(m0 + arow) * D_MODEL + k0 + ach;
        async_cp16(&As[bi][arow * 40 + ach],     gA);
        async_cp16(&As[bi][arow * 40 + ach + 8], gA + 8);
        const unsigned short* gB = Wt + (size_t)(k0 + bk) * D_MODEL + n0 + bc;
        async_cp16(&Bs[bi][bk * 136 + bc],     gB);
        async_cp16(&Bs[bi][bk * 136 + bc + 8], gB + 8);
    };

    const int NT = D_MODEL / 32;  // 32 K-tiles
    stage(0, 0);                  // prologue

    for (int t = 0; t < NT; t++) {
        if (t + 1 < NT) {
            stage((t + 1) * 32, (t + 1) & 1);  // prefetch next tile
            wait_async<4>();                   // older tile resident (in-order)
        } else {
            wait_async<0>();
        }
        __syncthreads();  // all waves' tile-t asyncs visible

        const unsigned short* as = As[t & 1];
        const unsigned short* bs = Bs[t & 1];

        // A fragments (16x32): lane m = mlo, half select by hi
        v16bf a[2];
#pragma unroll
        for (int mf = 0; mf < 2; mf++) {
            const int m = wm * 32 + mf * 16 + mlo;
            a[mf] = ld_frag2(&as[m * 40 + hi * 8], &as[m * 40 + 16 + hi * 8]);
        }
        // B fragments (32x16): lane n = mlo, elements k in table order
        const int kb = mlo + hi * 16;
#pragma unroll
        for (int nf = 0; nf < 4; nf++) {
            const int n = wn * 64 + nf * 16;
            v16u bu;
#pragma unroll
            for (int s = 0; s < 8; s++) {
                bu[s]     = bs[(hi * 8 + s) * 136 + n + mlo];
                bu[8 + s] = bs[(16 + hi * 8 + s) * 136 + n + mlo];
            }
            v16bf b = __builtin_bit_cast(v16bf, bu);
#pragma unroll
            for (int mf = 0; mf < 2; mf++)
                c[mf][nf] = __builtin_amdgcn_wmma_f32_16x16x32_bf16(
                    false, a[mf], false, b, (short)0, c[mf][nf], false, false);
        }
        __syncthreads();  // reads done before this buffer is overwritten
    }

    // store: D layout lane n = mlo, rows = hi*8 + r
#pragma unroll
    for (int mf = 0; mf < 2; mf++) {
#pragma unroll
        for (int nf = 0; nf < 4; nf++) {
            const int col = n0 + wn * 64 + nf * 16 + mlo;
            const float bv = bias[col];
#pragma unroll
            for (int r = 0; r < 8; r++) {
                const int row = m0 + wm * 32 + mf * 16 + hi * 8 + r;
                const float v = c[mf][nf][r] + bv;
                if (OUT_BF16)
                    ((unsigned short*)Optr)[(size_t)row * D_MODEL + col] = f2bf(v);
                else
                    ((float*)Optr)[(size_t)row * D_MODEL + col] = v;
            }
        }
    }
}

// ---------- Flash attention, one wave per 16-query tile ----------
// S^T = K * Q^T so that the score D-frag layout equals the A-frag layout
// needed for P*V (no cross-lane movement for P). 2 waves/block, per-wave
// double-buffered K/V tiles streamed with async global->LDS.

__global__ void __launch_bounds__(64)
attn_causal(const unsigned short* __restrict__ Qb,
            const unsigned short* __restrict__ Kb,
            const unsigned short* __restrict__ Vb,
            unsigned short* __restrict__ Ctx) {
    __shared__ __align__(16) unsigned short sQt[2][DK * 16];     // Q^T per wave
    __shared__ __align__(16) unsigned short sK[2][2][32 * DK];   // double buf
    __shared__ __align__(16) unsigned short sV[2][2][32 * DK];

    const int lane = threadIdx.x & 31;
    const int wave = threadIdx.x >> 5;
    const int q0   = (blockIdx.x * 2 + wave) * 16;
    const int b    = blockIdx.y >> 4;
    const int h    = blockIdx.y & 15;
    const size_t base = (size_t)b * SEQ * D_MODEL + (size_t)h * DK;

    unsigned short* Qt = sQt[wave];

    const int nlo = lane & 15;
    const int hi  = lane >> 4;

    auto stage_kv = [&](int j0, int bi) {
        const unsigned short* ksrc = Kb + base + (size_t)(j0 + lane) * D_MODEL;
        const unsigned short* vsrc = Vb + base + (size_t)(j0 + lane) * D_MODEL;
        unsigned short* kd = sK[wave][bi] + lane * DK;
        unsigned short* vd = sV[wave][bi] + lane * DK;
#pragma unroll
        for (int i = 0; i < 8; i++) {
            async_cp16(kd + 8 * i, ksrc + 8 * i);
            async_cp16(vd + 8 * i, vsrc + 8 * i);
        }
    };

    // stage Q^T: Qt[d][q] ; lane handles row q=lane>>1, 32 features
    {
        const int q  = lane >> 1;
        const int dh = (lane & 1) * 32;
        const unsigned short* src = Qb + base + (size_t)(q0 + q) * D_MODEL + dh;
#pragma unroll
        for (int i = 0; i < 4; i++) {
            uint4 v = *(const uint4*)(src + 8 * i);
            unsigned int w[4] = {v.x, v.y, v.z, v.w};
#pragma unroll
            for (int j = 0; j < 4; j++) {
                const int d = dh + 8 * i + 2 * j;
                Qt[(d + 0) * 16 + q] = (unsigned short)(w[j] & 0xFFFFu);
                Qt[(d + 1) * 16 + q] = (unsigned short)(w[j] >> 16);
            }
        }
    }

    stage_kv(0, 0);  // prologue: first K/V tile in flight

    // loop-invariant B-frags of Q^T (d-halves 0..31, 32..63)
    v16bf bq[2];
#pragma unroll
    for (int dh = 0; dh < 2; dh++) {
        const unsigned short* qr = Qb + base + (size_t)(q0 + nlo) * D_MODEL + dh * 32;
        bq[dh] = ld_frag2(qr + hi * 8, qr + 16 + hi * 8);
    }

    float m_run = -1e30f, l_run = 0.0f;
    v8f ctx[4];
#pragma unroll
    for (int f = 0; f < 4; f++) ctx[f] = zero8();

    const int   qg    = q0 + nlo;  // this lane's query index in S^T frags
    const float scale = 0.125f;    // 1/sqrt(64)
    const int   ktmax = (q0 + 15) >> 5;

    for (int kt = 0; kt <= ktmax; kt++) {
        const int j0 = kt * 32;
        if (kt < ktmax) {
            wait_ds0();                       // prior reads of buf^1 retired
            stage_kv(j0 + 32, (kt + 1) & 1);  // prefetch next K/V tile
            wait_async<16>();                 // current tile resident (in-order)
        } else {
            wait_async<0>();
        }
        const unsigned short* Ks = sK[wave][kt & 1];
        const unsigned short* Vs = sV[wave][kt & 1];

        // S^T (32 keys x 16 q) as two 16x16 frags; K-dim = features (2x32)
        v8f st[2];
        st[0] = zero8(); st[1] = zero8();
#pragma unroll
        for (int kh = 0; kh < 2; kh++) {
            const int key = kh * 16 + nlo;
#pragma unroll
            for (int dh = 0; dh < 2; dh++) {
                v16bf ak = ld_frag2(&Ks[key * DK + dh * 32 + hi * 8],
                                    &Ks[key * DK + dh * 32 + 16 + hi * 8]);
                st[kh] = __builtin_amdgcn_wmma_f32_16x16x32_bf16(
                    false, ak, false, bq[dh], (short)0, st[kh], false, false);
            }
        }

        // scale + causal mask (element: key j0+kh*16+hi*8+r vs query qg)
#pragma unroll
        for (int kh = 0; kh < 2; kh++)
#pragma unroll
            for (int r = 0; r < 8; r++) {
                float s = st[kh][r] * scale;
                const int keyg = j0 + kh * 16 + hi * 8 + r;
                if (keyg > qg) s = -1e30f;
                st[kh][r] = s;
            }

        // online softmax (per query = per lane pair {nlo, nlo+16})
        float mt = -1e30f;
#pragma unroll
        for (int kh = 0; kh < 2; kh++)
#pragma unroll
            for (int r = 0; r < 8; r++) mt = fmaxf(mt, st[kh][r]);
        mt = fmaxf(mt, __shfl_xor(mt, 16, 32));
        const float m_new = fmaxf(m_run, mt);
        const float alpha = __expf(m_run - m_new);

        float ps = 0.0f;
#pragma unroll
        for (int kh = 0; kh < 2; kh++)
#pragma unroll
            for (int r = 0; r < 8; r++) {
                const float p = __expf(st[kh][r] - m_new);
                st[kh][r] = p;
                ps += p;
            }
        ps += __shfl_xor(ps, 16, 32);
        l_run = l_run * alpha + ps;
        m_run = m_new;

        // rescale ctx rows: row q (local) = hi*8 + r, alpha lives in lane q
        float a_sh[8];
#pragma unroll
        for (int r = 0; r < 8; r++) a_sh[r] = __shfl(alpha, hi * 8 + r, 32);
#pragma unroll
        for (int f = 0; f < 4; f++)
#pragma unroll
            for (int r = 0; r < 8; r++) ctx[f][r] *= a_sh[r];

        // P: D-layout of S^T maps directly to A-frag (16q x 32key)
        v16u pu;
#pragma unroll
        for (int t = 0; t < 8; t++) {
            pu[t]     = f2bf(st[0][t]);
            pu[8 + t] = f2bf(st[1][t]);
        }
        const v16bf pa = __builtin_bit_cast(v16bf, pu);

        // ctx += P * V ; V B-frags per lane d = nlo, keys in table order
        const int kb = nlo + hi * 16;
#pragma unroll
        for (int nf = 0; nf < 4; nf++) {
            v16u vu;
#pragma unroll
            for (int s = 0; s < 8; s++) {
                vu[s]     = Vs[(hi * 8 + s) * DK + nf * 16 + nlo];
                vu[8 + s] = Vs[(16 + hi * 8 + s) * DK + nf * 16 + nlo];
            }
            v16bf bv = __builtin_bit_cast(v16bf, vu);
            ctx[nf] = __builtin_amdgcn_wmma_f32_16x16x32_bf16(
                false, pa, false, bv, (short)0, ctx[nf], false, false);
        }
    }

    // epilogue: divide by l, store bf16 ctx
    float linv[8];
#pragma unroll
    for (int r = 0; r < 8; r++) {
        const float lr = __shfl(l_run, hi * 8 + r, 32);
        linv[r] = 1.0f / lr;
    }
#pragma unroll
    for (int nf = 0; nf < 4; nf++) {
        const int d = nf * 16 + nlo;
#pragma unroll
        for (int r = 0; r < 8; r++) {
            const int row = q0 + hi * 8 + r;
            Ctx[base + (size_t)row * D_MODEL + d] = f2bf(ctx[nf][r] * linv[r]);
        }
    }
}

// ---------- launch ----------

extern "C" void kernel_launch(void* const* d_in, const int* in_sizes, int n_in,
                              void* d_out, int out_size, void* d_ws, size_t ws_size,
                              hipStream_t stream) {
    (void)in_sizes; (void)n_in; (void)out_size; (void)ws_size;

    const float* x  = (const float*)d_in[0];
    const float* Wq = (const float*)d_in[1];
    const float* bq = (const float*)d_in[2];
    const float* Wk = (const float*)d_in[3];
    const float* bk = (const float*)d_in[4];
    const float* Wv = (const float*)d_in[5];
    const float* bv = (const float*)d_in[6];
    const float* Wo = (const float*)d_in[7];
    const float* bo = (const float*)d_in[8];
    float* out = (float*)d_out;

    const size_t tok_elems = (size_t)NTOK * D_MODEL;      // 8M
    const size_t w_elems   = (size_t)D_MODEL * D_MODEL;   // 1M
    unsigned short* xb  = (unsigned short*)d_ws;
    unsigned short* Wqt = xb + tok_elems;
    unsigned short* Wkt = Wqt + w_elems;
    unsigned short* Wvt = Wkt + w_elems;
    unsigned short* Wot = Wvt + w_elems;
    unsigned short* Qb  = Wot + w_elems;
    unsigned short* Kb  = Qb + tok_elems;
    unsigned short* Vb  = Kb + tok_elems;
    unsigned short* Cb  = Vb + tok_elems;

    // pre-pass: convert x, transpose+convert weights
    cvt_bf16<<<(int)(tok_elems / 1024), 256, 0, stream>>>(x, xb, (int)tok_elems);
    dim3 tg(D_MODEL / 32, D_MODEL / 32);
    transpose_cvt<<<tg, 256, 0, stream>>>(Wq, Wqt);
    transpose_cvt<<<tg, 256, 0, stream>>>(Wk, Wkt);
    transpose_cvt<<<tg, 256, 0, stream>>>(Wv, Wvt);
    transpose_cvt<<<tg, 256, 0, stream>>>(Wo, Wot);

    dim3 gg(D_MODEL / 128, NTOK / 128);  // (8, 64)
    gemm_bias<true><<<gg, 256, 0, stream>>>(xb, Wqt, bq, Qb);
    gemm_bias<true><<<gg, 256, 0, stream>>>(xb, Wkt, bk, Kb);
    gemm_bias<true><<<gg, 256, 0, stream>>>(xb, Wvt, bv, Vb);

    attn_causal<<<dim3(SEQ / 32, BATCH * N_HEADS), 64, 0, stream>>>(Qb, Kb, Vb, Cb);

    gemm_bias<false><<<gg, 256, 0, stream>>>(Cb, Wot, bo, out);
}
